// DeformableTransformer_5145370820701
// MI455X (gfx1250) — compile-verified, matrix operand order checked by reading, not run on hardware
//
#include <hip/hip_runtime.h>
#include <hip/hip_bf16.h>
#include <stdint.h>

// ---------------- CDNA5 WMMA types ----------------
typedef __attribute__((ext_vector_type(16))) __bf16 v16bf;
typedef __attribute__((ext_vector_type(8)))  float  v8f;

union Frag { v16bf v; uint32_t w[8]; };

static __host__ __device__ inline int cdiv_i(int a, int b) { return (a + b - 1) / b; }

// Problem constants
#define BSZ   4
#define NQn   300
#define Dm    256
#define NHh   8
#define DHh   32
#define DFFn  1024
#define LTn   17821
#define M1n   (BSZ * NQn)      // 1200
#define MVn   (BSZ * LTn)      // 71284
#define SKP   320              // padded key length for softmax / P@V (multiple of 32)

// =====================================================================
// Elementwise helpers
// =====================================================================
__global__ __launch_bounds__(256) void k_cvt(const float* __restrict__ a,
                                             __bf16* __restrict__ o, int n) {
  int i = blockIdx.x * 256 + threadIdx.x;
  if (i < n) o[i] = (__bf16)a[i];
}

__global__ __launch_bounds__(256) void k_add(const float* __restrict__ a,
                                             const float* __restrict__ b,
                                             float* __restrict__ c, int n) {
  int i = blockIdx.x * 256 + threadIdx.x;
  if (i < n) c[i] = a[i] + b[i];
}

__global__ __launch_bounds__(256) void k_add_cvt(const float* __restrict__ a,
                                                 const float* __restrict__ b,
                                                 __bf16* __restrict__ o, int n) {
  int i = blockIdx.x * 256 + threadIdx.x;
  if (i < n) o[i] = (__bf16)(a[i] + b[i]);
}

// LayerNorm over 256 cols; one wave32 per row, 8 elems / lane.
// If B != nullptr the input is A+B (fused residual).
__global__ __launch_bounds__(256) void k_add_ln(const float* __restrict__ A,
                                                const float* __restrict__ B,
                                                const float* __restrict__ g,
                                                const float* __restrict__ be,
                                                float* __restrict__ out, int rows) {
  int row = (blockIdx.x * 256 + threadIdx.x) >> 5;
  if (row >= rows) return;
  int lane = threadIdx.x & 31;
  const float* a = A + (size_t)row * Dm;
  float x[8]; float s = 0.f;
  #pragma unroll
  for (int i = 0; i < 8; i++) {
    int c = lane + i * 32;
    x[i] = a[c] + (B ? B[(size_t)row * Dm + c] : 0.f);
    s += x[i];
  }
  #pragma unroll
  for (int o = 16; o > 0; o >>= 1) s += __shfl_xor(s, o, 32);
  float mean = s * (1.f / Dm);
  float vs = 0.f;
  #pragma unroll
  for (int i = 0; i < 8; i++) { float d = x[i] - mean; vs += d * d; }
  #pragma unroll
  for (int o = 16; o > 0; o >>= 1) vs += __shfl_xor(vs, o, 32);
  float inv = rsqrtf(vs * (1.f / Dm) + 1e-5f);
  float* op = out + (size_t)row * Dm;
  #pragma unroll
  for (int i = 0; i < 8; i++) {
    int c = lane + i * 32;
    op[c] = (x[i] - mean) * inv * g[c] + be[c];
  }
}

// =====================================================================
// Generic GEMM: C[M,N] = A[M,K](bf16) @ W[N,K]^T(bf16) + bias, f32 out.
// One wave computes a 16x32 tile: 1 A-frag, 2 B-frags, 2 WMMAs / K-step.
// N multiple of 32, K multiple of 32, M arbitrary.
// =====================================================================
template <int RELU>
__global__ __launch_bounds__(256) void k_gemm(const __bf16* __restrict__ A, int lda,
                                              const __bf16* __restrict__ W, int ldw,
                                              const float* __restrict__ bias,
                                              float* __restrict__ C, int ldc,
                                              int M, int N, int K) {
  const int lane = threadIdx.x & 31;
  const int wid  = (blockIdx.x * 256 + threadIdx.x) >> 5;
  const int tilesN = N >> 5;
  const int tilesM = (M + 15) >> 4;
  if (wid >= tilesM * tilesN) return;
  const int tm = wid / tilesN;
  const int tn = wid % tilesN;

  const int nl = lane & 15;
  const int hi = lane >> 4;
  const int row = tm * 16 + nl;
  const bool rowOK = (row < M);

  const int n0 = tn * 32 + nl;
  const int n1 = n0 + 16;

  v8f c0, c1;
  const float b0 = bias ? bias[n0] : 0.f;
  const float b1 = bias ? bias[n1] : 0.f;
  #pragma unroll
  for (int j = 0; j < 8; j++) { c0[j] = b0; c1[j] = b1; }

  const __bf16* aRow = A + (size_t)row * lda;
  const __bf16* w0p  = W + (size_t)n0 * ldw;
  const __bf16* w1p  = W + (size_t)n1 * ldw;

  for (int k0 = 0; k0 < K; k0 += 32) {
    Frag a, bA, bB;
    if (rowOK) {
      *(uint4*)&a.w[0] = *(const uint4*)(aRow + k0 + hi * 8);
      *(uint4*)&a.w[4] = *(const uint4*)(aRow + k0 + 16 + hi * 8);
      __builtin_prefetch(aRow + k0 + 64, 0, 0);   // global_prefetch next A chunk
    } else {
      #pragma unroll
      for (int j = 0; j < 8; j++) a.w[j] = 0u;
    }
    *(uint4*)&bA.w[0] = *(const uint4*)(w0p + k0 + hi * 8);
    *(uint4*)&bA.w[4] = *(const uint4*)(w0p + k0 + 16 + hi * 8);
    *(uint4*)&bB.w[0] = *(const uint4*)(w1p + k0 + hi * 8);
    *(uint4*)&bB.w[4] = *(const uint4*)(w1p + k0 + 16 + hi * 8);
    c0 = __builtin_amdgcn_wmma_f32_16x16x32_bf16(false, a.v, false, bA.v, (short)0, c0, false, false);
    c1 = __builtin_amdgcn_wmma_f32_16x16x32_bf16(false, a.v, false, bB.v, (short)0, c1, false, false);
  }

  #pragma unroll
  for (int j = 0; j < 8; j++) {
    int r = tm * 16 + j + 8 * hi;
    if (r < M) {
      float x0 = c0[j], x1 = c1[j];
      if (RELU) { x0 = fmaxf(x0, 0.f); x1 = fmaxf(x1, 0.f); }
      C[(size_t)r * ldc + n0] = x0;
      C[(size_t)r * ldc + n1] = x1;
    }
  }
}

// =====================================================================
// Self-attention: scores S[b,h,q,k] = Qh . Kh / sqrt(32)  (K=32 -> 1 WMMA)
// qkv layout: [B*NQ, 768] bf16, cols 0:256=Q, 256:512=K, 512:768=V.
// S stored [B,NH,300,320]; cols >=300 get -1e30 for softmax masking.
// =====================================================================
__global__ __launch_bounds__(256) void k_scores(const __bf16* __restrict__ qkv,
                                                float* __restrict__ S) {
  const int lane = threadIdx.x & 31;
  int wid = (blockIdx.x * 256 + threadIdx.x) >> 5;
  const int tilesM = 19, tilesN = 20, perBH = tilesM * tilesN;
  if (wid >= BSZ * NHh * perBH) return;
  int bh = wid / perBH, t = wid % perBH;
  int b = bh >> 3, h = bh & 7;
  int tm = t / tilesN, tn = t % tilesN;
  int nl = lane & 15, hi = lane >> 4;
  int qrow = tm * 16 + nl;
  int krow = tn * 16 + nl;

  const __bf16* qbase = qkv + (size_t)(b * NQn) * 768 + h * DHh;
  Frag a, bK;
  if (qrow < NQn) {
    const __bf16* p = qbase + (size_t)qrow * 768;
    *(uint4*)&a.w[0] = *(const uint4*)(p + hi * 8);
    *(uint4*)&a.w[4] = *(const uint4*)(p + 16 + hi * 8);
  } else {
    #pragma unroll
    for (int j = 0; j < 8; j++) a.w[j] = 0u;
  }
  if (krow < NQn) {
    const __bf16* p = qbase + 256 + (size_t)krow * 768;
    *(uint4*)&bK.w[0] = *(const uint4*)(p + hi * 8);
    *(uint4*)&bK.w[4] = *(const uint4*)(p + 16 + hi * 8);
  } else {
    #pragma unroll
    for (int j = 0; j < 8; j++) bK.w[j] = 0u;
  }

  v8f c;
  #pragma unroll
  for (int j = 0; j < 8; j++) c[j] = 0.f;
  c = __builtin_amdgcn_wmma_f32_16x16x32_bf16(false, a.v, false, bK.v, (short)0, c, false, false);

  const float scale = 0.17677669529663687f;  // 1/sqrt(32)
  float* Sb = S + ((size_t)bh * NQn) * SKP;
  int n = tn * 16 + nl;
  #pragma unroll
  for (int j = 0; j < 8; j++) {
    int r = tm * 16 + j + 8 * hi;
    if (r < NQn) Sb[(size_t)r * SKP + n] = (n < NQn) ? c[j] * scale : -1e30f;
  }
}

// softmax over 320 padded cols (one wave32 per row), write bf16 P
__global__ __launch_bounds__(256) void k_softmax(const float* __restrict__ S,
                                                 __bf16* __restrict__ P) {
  int row = (blockIdx.x * 256 + threadIdx.x) >> 5;
  if (row >= BSZ * NHh * NQn) return;
  int lane = threadIdx.x & 31;
  const float* s = S + (size_t)row * SKP;
  float v[10], mx = -1e30f;
  #pragma unroll
  for (int i = 0; i < 10; i++) { v[i] = s[lane + i * 32]; mx = fmaxf(mx, v[i]); }
  #pragma unroll
  for (int o = 16; o > 0; o >>= 1) mx = fmaxf(mx, __shfl_xor(mx, o, 32));
  float sum = 0.f;
  #pragma unroll
  for (int i = 0; i < 10; i++) { v[i] = __expf(v[i] - mx); sum += v[i]; }
  #pragma unroll
  for (int o = 16; o > 0; o >>= 1) sum += __shfl_xor(sum, o, 32);
  float inv = 1.f / sum;
  __bf16* p = P + (size_t)row * SKP;
  #pragma unroll
  for (int i = 0; i < 10; i++) p[lane + i * 32] = (__bf16)(v[i] * inv);
}

// O[b,q,h*32+d] = sum_k P[b,h,q,k] * V[b,k,h,d]   (WMMA, K loop over 320)
__global__ __launch_bounds__(256) void k_attn_o(const __bf16* __restrict__ P,
                                                const __bf16* __restrict__ qkv,
                                                float* __restrict__ O) {
  const int lane = threadIdx.x & 31;
  int wid = (blockIdx.x * 256 + threadIdx.x) >> 5;
  const int tilesM = 19;
  if (wid >= BSZ * NHh * tilesM) return;
  int bh = wid / tilesM, tm = wid % tilesM;
  int b = bh >> 3, h = bh & 7;
  int nl = lane & 15, hi = lane >> 4;
  int qrow = tm * 16 + nl;

  const __bf16* prow  = P + ((size_t)bh * NQn) * SKP;
  const __bf16* vbase = qkv + (size_t)b * NQn * 768 + 512 + h * DHh;
  const __bf16 z = (__bf16)0.f;

  v8f c0, c1;
  #pragma unroll
  for (int j = 0; j < 8; j++) { c0[j] = 0.f; c1[j] = 0.f; }

  for (int k0 = 0; k0 < SKP; k0 += 32) {
    Frag a, b0f, b1f;
    if (qrow < NQn) {
      const __bf16* p = prow + (size_t)qrow * SKP + k0;
      *(uint4*)&a.w[0] = *(const uint4*)(p + hi * 8);
      *(uint4*)&a.w[4] = *(const uint4*)(p + 16 + hi * 8);
    } else {
      #pragma unroll
      for (int j = 0; j < 8; j++) a.w[j] = 0u;
    }
    #pragma unroll
    for (int i = 0; i < 16; i++) {
      int k = k0 + ((i < 8) ? (hi * 8 + i) : (16 + hi * 8 + (i - 8)));
      __bf16 e0 = z, e1 = z;
      if (k < NQn) { e0 = vbase[(size_t)k * 768 + nl];
                     e1 = vbase[(size_t)k * 768 + 16 + nl]; }
      b0f.v[i] = e0; b1f.v[i] = e1;
    }
    c0 = __builtin_amdgcn_wmma_f32_16x16x32_bf16(false, a.v, false, b0f.v, (short)0, c0, false, false);
    c1 = __builtin_amdgcn_wmma_f32_16x16x32_bf16(false, a.v, false, b1f.v, (short)0, c1, false, false);
  }

  float* obase = O + (size_t)b * NQn * Dm + h * DHh;
  #pragma unroll
  for (int j = 0; j < 8; j++) {
    int r = tm * 16 + j + 8 * hi;
    if (r < NQn) {
      obase[(size_t)r * Dm + nl]      = c0[j];
      obase[(size_t)r * Dm + 16 + nl] = c1[j];
    }
  }
}

// =====================================================================
// Deformable attention prep: softmax attention weights + sampling locations
// one thread per (b,q,h)
// =====================================================================
__global__ __launch_bounds__(256) void k_prep(const float* __restrict__ off,
                                              const float* __restrict__ awl,
                                              const float* __restrict__ ref,
                                              float* __restrict__ loc,
                                              float* __restrict__ awn) {
  int t = blockIdx.x * 256 + threadIdx.x;
  if (t >= BSZ * NQn * NHh) return;
  int h = t & 7;
  int bq = t >> 3;                // b*300 + q
  const float* al = awl + (size_t)bq * 128 + h * 16;
  float e[16], mx = -1e30f;
  #pragma unroll
  for (int i = 0; i < 16; i++) mx = fmaxf(mx, al[i]);
  float s = 0.f;
  #pragma unroll
  for (int i = 0; i < 16; i++) { e[i] = __expf(al[i] - mx); s += e[i]; }
  float inv = 1.f / s;

  const float invW[4] = {1.f/134.f, 1.f/67.f, 1.f/34.f, 1.f/17.f};
  const float invH[4] = {1.f/100.f, 1.f/50.f, 1.f/25.f, 1.f/13.f};
  const float* of = off + (size_t)bq * Dm + h * 32;  // [NL,NP,2]
  const float* rf = ref + (size_t)bq * 8;            // [NL,2]
  float* lc = loc + (size_t)t * 32;
  float* an = awn + (size_t)t * 16;
  #pragma unroll
  for (int l = 0; l < 4; l++) {
    #pragma unroll
    for (int p = 0; p < 4; p++) {
      int i = l * 4 + p;
      an[i] = e[i] * inv;
      lc[i * 2 + 0] = rf[l * 2 + 0] + of[i * 2 + 0] * invW[l];
      lc[i * 2 + 1] = rf[l * 2 + 1] + of[i * 2 + 1] * invH[l];
    }
  }
}

// Bilinear sampling + weighted sum; one thread per (b,q,h,d); lanes = d
// (coalesced 32-float gathers; 73MB value tensor lives in the 192MB L2)
__global__ __launch_bounds__(256) void k_sample(const float* __restrict__ val,
                                                const float* __restrict__ loc,
                                                const float* __restrict__ awn,
                                                float* __restrict__ out) {
  int t = blockIdx.x * 256 + threadIdx.x;
  if (t >= BSZ * NQn * NHh * DHh) return;
  int d  = t & 31;
  int h  = (t >> 5) & 7;
  int bq = t >> 8;
  int b  = bq / NQn;
  int idx = bq * 8 + h;
  const float* lc = loc + (size_t)idx * 32;
  const float* an = awn + (size_t)idx * 16;
  const int Hs[4] = {100, 50, 25, 13};
  const int Ws[4] = {134, 67, 34, 17};
  const int ST[4] = {0, 13400, 16750, 17600};
  const float* vb = val + (size_t)b * LTn * Dm + h * DHh + d;

  float acc = 0.f;
  #pragma unroll
  for (int l = 0; l < 4; l++) {
    const int H = Hs[l], W = Ws[l], st = ST[l];
    #pragma unroll
    for (int p = 0; p < 4; p++) {
      int i = l * 4 + p;
      float x = lc[i * 2 + 0] * (float)W - 0.5f;
      float y = lc[i * 2 + 1] * (float)H - 0.5f;
      float xf = floorf(x), yf = floorf(y);
      float wx = x - xf, wy = y - yf;
      int x0 = (int)xf, y0 = (int)yf;
      int x1 = x0 + 1,  y1 = y0 + 1;
      float v00 = 0.f, v10 = 0.f, v01 = 0.f, v11 = 0.f;
      bool ix0 = (unsigned)x0 < (unsigned)W, ix1 = (unsigned)x1 < (unsigned)W;
      bool iy0 = (unsigned)y0 < (unsigned)H, iy1 = (unsigned)y1 < (unsigned)H;
      if (ix0 && iy0) v00 = vb[(size_t)(st + y0 * W + x0) * Dm];
      if (ix1 && iy0) v10 = vb[(size_t)(st + y0 * W + x1) * Dm];
      if (ix0 && iy1) v01 = vb[(size_t)(st + y1 * W + x0) * Dm];
      if (ix1 && iy1) v11 = vb[(size_t)(st + y1 * W + x1) * Dm];
      float sres = v00 * (1.f - wx) * (1.f - wy) + v10 * wx * (1.f - wy)
                 + v01 * (1.f - wx) * wy         + v11 * wx * wy;
      acc += an[i] * sres;
    }
  }
  out[t] = acc;  // layout [b,q,h,d] == [B*NQ, 256] row-major
}

// =====================================================================
// Host orchestration
// =====================================================================
extern "C" void kernel_launch(void* const* d_in, const int* in_sizes, int n_in,
                              void* d_out_v, int out_size, void* d_ws, size_t ws_size,
                              hipStream_t stream) {
  const float* tgt[2] = {(const float*)d_in[0], (const float*)d_in[1]};
  const float* pos[2] = {(const float*)d_in[2], (const float*)d_in[3]};
  const float* ref[2] = {(const float*)d_in[4], (const float*)d_in[5]};
  const float* src[2] = {(const float*)d_in[6], (const float*)d_in[7]};
  const float* sa_in_w  = (const float*)d_in[8];
  const float* sa_in_b  = (const float*)d_in[9];
  const float* sa_out_w = (const float*)d_in[10];
  const float* sa_out_b = (const float*)d_in[11];
  const float* ln_w = (const float*)d_in[12];
  const float* ln_b = (const float*)d_in[13];
  const float* ffn_w1 = (const float*)d_in[14];
  const float* ffn_b1 = (const float*)d_in[15];
  const float* ffn_w2 = (const float*)d_in[16];
  const float* ffn_b2 = (const float*)d_in[17];
  const float* val_w = (const float*)d_in[18];
  const float* val_b = (const float*)d_in[19];
  const float* off_w = (const float*)d_in[20];
  const float* off_b = (const float*)d_in[21];
  const float* aw_w  = (const float*)d_in[22];
  const float* aw_b  = (const float*)d_in[23];
  const float* cout_w = (const float*)d_in[24];
  const float* cout_b = (const float*)d_in[25];
  float* out = (float*)d_out_v;
  (void)in_sizes; (void)n_in; (void)out_size; (void)ws_size;

  // bump allocator over workspace
  char* base = (char*)d_ws;
  size_t cur = 0;
  auto alloc = [&](size_t bytes) -> void* {
    void* p = base + cur;
    cur += (bytes + 255) & ~(size_t)255;
    return p;
  };

  // bf16 weight copies
  __bf16* sa_in_w_bf  = (__bf16*)alloc((size_t)2 * 768 * 256 * 2);
  __bf16* sa_out_w_bf = (__bf16*)alloc((size_t)2 * 256 * 256 * 2);
  __bf16* ffn_w1_bf   = (__bf16*)alloc((size_t)2 * 1024 * 256 * 2);
  __bf16* ffn_w2_bf   = (__bf16*)alloc((size_t)2 * 256 * 1024 * 2);
  __bf16* val_w_bf    = (__bf16*)alloc((size_t)2 * 256 * 256 * 2);
  __bf16* off_w_bf    = (__bf16*)alloc((size_t)2 * 256 * 256 * 2);
  __bf16* aw_w_bf     = (__bf16*)alloc((size_t)2 * 128 * 256 * 2);
  __bf16* cout_w_bf   = (__bf16*)alloc((size_t)2 * 256 * 256 * 2);

  __bf16* qbf[2]   = {(__bf16*)alloc((size_t)M1n * Dm * 2), (__bf16*)alloc((size_t)M1n * Dm * 2)};
  __bf16* tgtbf[2] = {(__bf16*)alloc((size_t)M1n * Dm * 2), (__bf16*)alloc((size_t)M1n * Dm * 2)};
  __bf16* qcbf[2]  = {(__bf16*)alloc((size_t)M1n * Dm * 2), (__bf16*)alloc((size_t)M1n * Dm * 2)};
  float*  tgt2[2]  = {(float*)alloc((size_t)M1n * Dm * 4), (float*)alloc((size_t)M1n * Dm * 4)};

  float*  qkv_f   = (float*)alloc((size_t)M1n * 768 * 4);
  __bf16* qkv_bf  = (__bf16*)alloc((size_t)M1n * 768 * 2);
  float*  Sbuf    = (float*)alloc((size_t)BSZ * NHh * NQn * SKP * 4);
  __bf16* Pbuf    = (__bf16*)alloc((size_t)BSZ * NHh * NQn * SKP * 2);
  float*  attn_f  = (float*)alloc((size_t)M1n * Dm * 4);
  __bf16* attn_bf = (__bf16*)alloc((size_t)M1n * Dm * 2);
  float*  sa_f    = (float*)alloc((size_t)M1n * Dm * 4);

  __bf16* srcbf  = (__bf16*)alloc((size_t)MVn * Dm * 2);
  float*  value  = (float*)alloc((size_t)MVn * Dm * 4);
  float*  off_f  = (float*)alloc((size_t)M1n * Dm * 4);
  float*  awlog  = (float*)alloc((size_t)M1n * 128 * 4);
  float*  locb   = (float*)alloc((size_t)M1n * NHh * 32 * 4);
  float*  awnb   = (float*)alloc((size_t)M1n * NHh * 16 * 4);
  float*  sampf  = (float*)alloc((size_t)M1n * Dm * 4);
  __bf16* sampbf = (__bf16*)alloc((size_t)M1n * Dm * 2);

  float* rbuf = (float*)alloc((size_t)M1n * Dm * 4);
  float* tbuf = (float*)alloc((size_t)M1n * Dm * 4);
  float* xln[2] = {(float*)alloc((size_t)M1n * Dm * 4), (float*)alloc((size_t)M1n * Dm * 4)};
  __bf16* xbf  = (__bf16*)alloc((size_t)M1n * Dm * 2);
  float*  h1f  = (float*)alloc((size_t)M1n * DFFn * 4);
  __bf16* h1bf = (__bf16*)alloc((size_t)M1n * DFFn * 2);
  float*  h2f  = (float*)alloc((size_t)M1n * Dm * 4);

  auto cvt = [&](const float* a, __bf16* o, int n) {
    k_cvt<<<cdiv_i(n, 256), 256, 0, stream>>>(a, o, n);
  };
  auto gemm = [&](const __bf16* A, int lda, const __bf16* W, int ldw,
                  const float* bias, float* C, int ldc, int M, int N, int K, bool relu) {
    int waves = cdiv_i(M, 16) * (N / 32);
    int blocks = cdiv_i(waves, 8);
    if (relu) k_gemm<1><<<blocks, 256, 0, stream>>>(A, lda, W, ldw, bias, C, ldc, M, N, K);
    else      k_gemm<0><<<blocks, 256, 0, stream>>>(A, lda, W, ldw, bias, C, ldc, M, N, K);
  };

  // ---- convert weights to bf16 ----
  cvt(sa_in_w,  sa_in_w_bf,  2 * 768 * 256);
  cvt(sa_out_w, sa_out_w_bf, 2 * 256 * 256);
  cvt(ffn_w1,   ffn_w1_bf,   2 * 1024 * 256);
  cvt(ffn_w2,   ffn_w2_bf,   2 * 256 * 1024);
  cvt(val_w,    val_w_bf,    2 * 256 * 256);
  cvt(off_w,    off_w_bf,    2 * 256 * 256);
  cvt(aw_w,     aw_w_bf,     2 * 128 * 256);
  cvt(cout_w,   cout_w_bf,   2 * 256 * 256);

  // ---- dual self-attention + norm2 ----
  for (int m = 0; m < 2; m++) {
    k_add_cvt<<<cdiv_i(M1n * Dm, 256), 256, 0, stream>>>(tgt[m], pos[m], qbf[m], M1n * Dm);
    cvt(tgt[m], tgtbf[m], M1n * Dm);
    // QK projection (wq|wk): q = tgt + pos
    gemm(qbf[m], Dm, sa_in_w_bf + (size_t)m * 768 * 256, Dm,
         sa_in_b + (size_t)m * 768, qkv_f, 768, M1n, 512, Dm, false);
    // V projection: input = tgt (no pos)
    gemm(tgtbf[m], Dm, sa_in_w_bf + (size_t)m * 768 * 256 + 512 * 256, Dm,
         sa_in_b + (size_t)m * 768 + 512, qkv_f + 512, 768, M1n, 256, Dm, false);
    cvt(qkv_f, qkv_bf, M1n * 768);
    k_scores<<<cdiv_i(BSZ * NHh * 19 * 20, 8), 256, 0, stream>>>(qkv_bf, Sbuf);
    k_softmax<<<cdiv_i(BSZ * NHh * NQn, 8), 256, 0, stream>>>(Sbuf, Pbuf);
    k_attn_o<<<cdiv_i(BSZ * NHh * 19, 8), 256, 0, stream>>>(Pbuf, qkv_bf, attn_f);
    cvt(attn_f, attn_bf, M1n * Dm);
    gemm(attn_bf, Dm, sa_out_w_bf + (size_t)m * 256 * 256, Dm,
         sa_out_b + (size_t)m * 256, sa_f, Dm, M1n, Dm, Dm, false);
    // tgt2 = LN(tgt + sa)  [norm2]
    k_add_ln<<<cdiv_i(M1n, 8), 256, 0, stream>>>(tgt[m], sa_f,
        ln_w + (size_t)(m * 3 + 1) * Dm, ln_b + (size_t)(m * 3 + 1) * Dm, tgt2[m], M1n);
    // qc = tgt2 + pos (bf16)
    k_add_cvt<<<cdiv_i(M1n * Dm, 256), 256, 0, stream>>>(tgt2[m], pos[m], qcbf[m], M1n * Dm);
  }

  // ---- interactive deformable cross-attention (modality-swapped memory) ----
  for (int m = 0; m < 2; m++) {
    const float* src_other = src[1 - m];  // RGB queries sample T memory & vice versa
    cvt(src_other, srcbf, MVn * Dm);
    // value projection: the big GEMM (71284 x 256 x 256)
    gemm(srcbf, Dm, val_w_bf + (size_t)m * 256 * 256, Dm,
         val_b + (size_t)m * 256, value, Dm, MVn, Dm, Dm, false);
    // sampling offsets & attention-weight logits
    gemm(qcbf[m], Dm, off_w_bf + (size_t)m * 256 * 256, Dm,
         off_b + (size_t)m * 256, off_f, Dm, M1n, Dm, Dm, false);
    gemm(qcbf[m], Dm, aw_w_bf + (size_t)m * 128 * 256, Dm,
         aw_b + (size_t)m * 128, awlog, 128, M1n, 128, Dm, false);
    k_prep<<<cdiv_i(M1n * NHh, 256), 256, 0, stream>>>(off_f, awlog, ref[m], locb, awnb);
    k_sample<<<cdiv_i(M1n * Dm, 256), 256, 0, stream>>>(value, locb, awnb, sampf);
    cvt(sampf, sampbf, M1n * Dm);
    // output projection -> F_RGB / F_T directly into d_out
    gemm(sampbf, Dm, cout_w_bf + (size_t)m * 256 * 256, Dm,
         cout_b + (size_t)m * 256, out + (size_t)(2 + m) * M1n * Dm, Dm, M1n, Dm, Dm, false);
  }

  // ---- cross residuals + norm1 (note the modality swap in the reference) ----
  float* t2R = out + (size_t)2 * M1n * Dm;
  float* t2T = out + (size_t)3 * M1n * Dm;
  k_add<<<cdiv_i(M1n * Dm, 256), 256, 0, stream>>>(tgt2[1], t2R, rbuf, M1n * Dm);  // r = tgt_T + t2_RGB
  k_add<<<cdiv_i(M1n * Dm, 256), 256, 0, stream>>>(rbuf, t2T, tbuf, M1n * Dm);     // t = r + t2_T
  k_add_ln<<<cdiv_i(M1n, 8), 256, 0, stream>>>(rbuf, nullptr, ln_w + 0 * Dm, ln_b + 0 * Dm, xln[0], M1n);
  k_add_ln<<<cdiv_i(M1n, 8), 256, 0, stream>>>(tbuf, nullptr, ln_w + 3 * Dm, ln_b + 3 * Dm, xln[1], M1n);

  // ---- FFNs + norm3 ----
  for (int m = 0; m < 2; m++) {
    cvt(xln[m], xbf, M1n * Dm);
    gemm(xbf, Dm, ffn_w1_bf + (size_t)m * 1024 * 256, Dm,
         ffn_b1 + (size_t)m * 1024, h1f, DFFn, M1n, DFFn, Dm, true);   // ReLU fused
    cvt(h1f, h1bf, M1n * DFFn);
    gemm(h1bf, DFFn, ffn_w2_bf + (size_t)m * 256 * 1024, DFFn,
         ffn_b2 + (size_t)m * 256, h2f, Dm, M1n, Dm, DFFn, false);
    k_add_ln<<<cdiv_i(M1n, 8), 256, 0, stream>>>(xln[m], h2f,
        ln_w + (size_t)(m * 3 + 2) * Dm, ln_b + (size_t)(m * 3 + 2) * Dm,
        out + (size_t)m * M1n * Dm, M1n);
  }
}
